// Magi1Attention_7524782703196
// MI455X (gfx1250) — matrix operand order repaired and checked by reading, not compiled, not run on hardware
//
#include <hip/hip_runtime.h>
#include <hip/hip_bf16.h>
#include <stdint.h>

typedef __bf16 bf16_t;
typedef __attribute__((ext_vector_type(16))) __bf16 v16bf;
typedef __attribute__((ext_vector_type(8)))  __bf16 v8bf;
typedef __attribute__((ext_vector_type(8)))  float  v8f;
typedef __attribute__((ext_vector_type(4)))  int    v4i;

static constexpr int Bc = 2, Sc = 2048, DIMc = 3072, Hc = 24, Dc = 128;

// Use the CDNA5 async global->LDS path when the toolchain exposes it.
#if defined(__has_builtin)
#if __has_builtin(__builtin_amdgcn_global_load_async_to_lds_b128) && \
    __has_builtin(__builtin_amdgcn_s_wait_asynccnt)
#define USE_ASYNC_LDS 1
#endif
#endif
#ifndef USE_ASYNC_LDS
#define USE_ASYNC_LDS 0
#endif

__device__ __forceinline__ bf16_t f2bf(float f) {
  unsigned u = __builtin_bit_cast(unsigned, f);
  unsigned r = u + 0x7FFFu + ((u >> 16) & 1u);   // round-to-nearest-even
  unsigned short h = (unsigned short)(r >> 16);
  return __builtin_bit_cast(bf16_t, h);
}

__device__ __forceinline__ v16bf cat16(v8bf lo, v8bf hi) {
  return __builtin_shufflevector(lo, hi, 0,1,2,3,4,5,6,7,8,9,10,11,12,13,14,15);
}
// 16 contiguous bf16 (B-fragment per lane: elems K = e + 16*half)
__device__ __forceinline__ v16bf load16(const bf16_t* p) {
  return cat16(*(const v8bf*)p, *(const v8bf*)(p + 8));
}
// two separated 8-bf16 chunks (A-fragment per lane: K = e+16*half, as callers pass p1 = p0+8)
__device__ __forceinline__ v16bf load8x2(const bf16_t* p0, const bf16_t* p1) {
  return cat16(*(const v8bf*)p0, *(const v8bf*)p1);
}
__device__ __forceinline__ v8f wmma_bf16(v16bf a, v16bf b, v8f c) {
  return __builtin_amdgcn_wmma_f32_16x16x32_bf16(false, a, false, b, (short)0, c,
                                                 false, false);
}
__device__ __forceinline__ void copy32B(bf16_t* dst, const bf16_t* src) {
  *(uint4*)dst           = *(const uint4*)src;
  *((uint4*)dst + 1)     = *((const uint4*)src + 1);
}

#if USE_ASYNC_LDS
// 16B per lane, global -> LDS, no VGPR landing; tracked by ASYNCcnt.
__device__ __forceinline__ void async_cp16(const bf16_t* g, bf16_t* l) {
  __builtin_amdgcn_global_load_async_to_lds_b128(
      (__attribute__((address_space(1))) v4i*)(uintptr_t)g,
      (__attribute__((address_space(3))) v4i*)(uint32_t)(uintptr_t)l, 0, 0);
}
#endif

// ---------------------------------------------------------------------------
// Pre-pass A: flat f32 -> bf16 convert (hidden states), 4 elems/thread.
// ---------------------------------------------------------------------------
__global__ __launch_bounds__(256)
void convert_bf16_kernel(const float* __restrict__ src, bf16_t* __restrict__ dst,
                         int n4)
{
  int i = blockIdx.x * 256 + threadIdx.x;
  if (i >= n4) return;
  float4 f = *(const float4*)(src + (size_t)i * 4);
  union { bf16_t h[4]; uint2 u; } o;
  o.h[0] = f2bf(f.x); o.h[1] = f2bf(f.y); o.h[2] = f2bf(f.z); o.h[3] = f2bf(f.w);
  *(uint2*)(dst + (size_t)i * 4) = o.u;
}

// ---------------------------------------------------------------------------
// Pre-pass B: transpose + convert  src[K][N] f32  ->  dst[N][K] bf16.
// ---------------------------------------------------------------------------
__global__ __launch_bounds__(256)
void transpose_convert_kernel(const float* __restrict__ src, bf16_t* __restrict__ dst,
                              int K, int N)
{
  __shared__ float tile[32][33];
  const int n0 = blockIdx.x * 32, k0 = blockIdx.y * 32;
  const int tx = threadIdx.x & 31, ty = threadIdx.x >> 5;   // 32 x 8
  #pragma unroll
  for (int r = ty; r < 32; r += 8)
    tile[r][tx] = src[(size_t)(k0 + r) * N + n0 + tx];
  __syncthreads();
  #pragma unroll
  for (int r = ty; r < 32; r += 8)
    dst[(size_t)(n0 + r) * K + k0 + tx] = f2bf(tile[tx][r]);
}

// ---------------------------------------------------------------------------
// Kernel 1: QKV projection + fp32 LayerNorm + RoPE.
// 4 waves, block tile 64x128; wave tile 32x64 (acc[2][4]); register
// double-buffered fragments straight from global (bf16, WMMA layouts).
// blockIdx.y: 0..23 -> Q head, 24 -> K, 25 -> V.
// ---------------------------------------------------------------------------
__global__ __launch_bounds__(128)
void qkv_ln_rope_kernel(const bf16_t* __restrict__ hb,
                        const bf16_t* __restrict__ wqT,
                        const bf16_t* __restrict__ wkT,
                        const bf16_t* __restrict__ wvT,
                        const float* __restrict__ cosv,
                        const float* __restrict__ sinv,
                        const float* __restrict__ ln_q_w, const float* __restrict__ ln_q_b,
                        const float* __restrict__ ln_k_w, const float* __restrict__ ln_k_b,
                        bf16_t* __restrict__ qb, bf16_t* __restrict__ kbuf,
                        bf16_t* __restrict__ vt)
{
  __shared__ __align__(16) float outb[64 * 128];   // fp32 result tile for LN
  __shared__ float muv[64], rsv[64];

  const int tid  = threadIdx.x;
  const int lane = tid & 31, wave = tid >> 5;
  const int half = lane >> 4, l16 = lane & 15;
  const int mi = wave >> 1, ni = wave & 1;         // wave: rows mi*32, cols ni*64
  const int m0 = blockIdx.x * 64;
  const int hy = blockIdx.y;

  const bf16_t* wT;
  const float *lw = nullptr, *lb = nullptr; bool lnr;
  if (hy < Hc)       { wT = wqT + (size_t)hy * Dc * DIMc; lw = ln_q_w; lb = ln_q_b; lnr = true;  }
  else if (hy == Hc) { wT = wkT;                          lw = ln_k_w; lb = ln_k_b; lnr = true;  }
  else               { wT = wvT;                                                    lnr = false; }

  v8f z = {};
  v8f acc[2][4];
  #pragma unroll
  for (int m = 0; m < 2; ++m)
    #pragma unroll
    for (int j = 0; j < 4; ++j) acc[m][j] = z;

  const bf16_t* arow[2];
  #pragma unroll
  for (int m = 0; m < 2; ++m)
    arow[m] = hb + (size_t)(m0 + mi * 32 + m * 16 + l16) * DIMc;
  const bf16_t* bcol[4];
  #pragma unroll
  for (int j = 0; j < 4; ++j)
    bcol[j] = wT + (size_t)(ni * 64 + j * 16 + l16) * DIMc + 16 * half;

  auto loadA = [&](v16bf af[2], int kb) {
    #pragma unroll
    for (int m = 0; m < 2; ++m)
      af[m] = load8x2(arow[m] + kb + 16 * half, arow[m] + kb + 16 * half + 8);
  };
  auto loadB = [&](v16bf bfv[4], int kb) {
    #pragma unroll
    for (int j = 0; j < 4; ++j) bfv[j] = load16(bcol[j] + kb);
  };
  auto mmas = [&](v16bf af[2], v16bf bfv[4]) {
    #pragma unroll
    for (int m = 0; m < 2; ++m)
      #pragma unroll
      for (int j = 0; j < 4; ++j)
        acc[m][j] = wmma_bf16(af[m], bfv[j], acc[m][j]);
  };

  v16bf a0[2], b0[4], a1[2], b1[4];
  loadA(a0, 0); loadB(b0, 0);
  for (int kb = 0; kb < DIMc; kb += 64) {          // 2 K-steps per iteration
    loadA(a1, kb + 32); loadB(b1, kb + 32);        // prefetch odd step
    mmas(a0, b0);
    if (kb + 64 < DIMc) { loadA(a0, kb + 64); loadB(b0, kb + 64); }
    mmas(a1, b1);
  }

  #pragma unroll
  for (int m = 0; m < 2; ++m)
    #pragma unroll
    for (int j = 0; j < 4; ++j)
      #pragma unroll
      for (int r = 0; r < 8; ++r)
        outb[(mi * 32 + m * 16 + r + 8 * half) * 128 + ni * 64 + j * 16 + l16] =
            acc[m][j][r];
  __syncthreads();

  if (tid < 64) { // per-row mean / rsigma over D=128
    float s = 0.f, ss = 0.f;
    #pragma unroll 4
    for (int c = 0; c < 128; ++c) { float v = outb[tid * 128 + c]; s += v; ss += v * v; }
    float mu = s * (1.f / 128.f);
    float var = ss * (1.f / 128.f) - mu * mu;
    muv[tid] = mu; rsv[tid] = rsqrtf(var + 1e-6f);
  }
  __syncthreads();

  #pragma unroll 2
  for (int p = 0; p < 32; ++p) {               // 4096 rope pairs / 128 threads
    int idx = tid * 32 + p;
    int row = idx >> 6, i = idx & 63;          // i = rope pair index (0..63)
    int token = m0 + row, bb = token >> 11, pos = token & (Sc - 1);
    float y1 = outb[row * 128 + 2 * i], y2 = outb[row * 128 + 2 * i + 1];
    float o1, o2;
    if (lnr) {
      float mu = muv[row], rs = rsv[row];
      float h1 = (y1 - mu) * rs * lw[2 * i]     + lb[2 * i];
      float h2 = (y2 - mu) * rs * lw[2 * i + 1] + lb[2 * i + 1];
      float c = cosv[pos * Dc + 2 * i];
      float s = sinv[pos * Dc + 2 * i + 1];
      o1 = h1 * c - h2 * s;
      o2 = h1 * s + h2 * c;
    } else { o1 = y1; o2 = y2; }
    if (hy < Hc) {
      bf16_t* d = qb + ((size_t)((bb * Hc + hy) * Sc + pos)) * Dc + 2 * i;
      d[0] = f2bf(o1); d[1] = f2bf(o2);
    } else if (hy == Hc) {
      bf16_t* d = kbuf + ((size_t)(bb * Sc + pos)) * Dc + 2 * i;
      d[0] = f2bf(o1); d[1] = f2bf(o2);
    } else { // V transposed: vt[b][d][s]
      vt[((size_t)(bb * Dc + 2 * i)) * Sc + pos]     = f2bf(o1);
      vt[((size_t)(bb * Dc + 2 * i + 1)) * Sc + pos] = f2bf(o2);
    }
  }
}

// ---------------------------------------------------------------------------
// Kernel 2: flash-style MQA attention. 8 waves x 16 query rows per block.
// K and V^T blocks staged cooperatively into double-buffered LDS (all waves
// share the same KV stream -> 8x less L2 traffic); one barrier per key block.
// Staging uses async global->LDS when available (ASYNCcnt), overlapping the
// next block's DMA with this block's WMMA + softmax.
// ---------------------------------------------------------------------------
__global__ __launch_bounds__(256)
void attention_kernel(const bf16_t* __restrict__ qb, const bf16_t* __restrict__ kbuf,
                      const bf16_t* __restrict__ vt, bf16_t* __restrict__ ao)
{
  __shared__ __align__(16) bf16_t kt[2][32 * 128];     // [key][d]
  __shared__ __align__(16) bf16_t vtile[2][128 * 32];  // [d][key]
  __shared__ __align__(16) bf16_t pbuf[8 * 512];       // per-wave P transpose
  const int tid = threadIdx.x, lane = tid & 31, wave = tid >> 5;
  const int half = lane >> 4, l16 = lane & 15;
  const int q0 = blockIdx.x * 128 + wave * 16;
  const int h = blockIdx.y, bb = blockIdx.z;
  bf16_t* pw = &pbuf[wave * 512];
  const float scale = 0.08838834764831845f;            // 1/sqrt(128)

  const bf16_t* kbase = kbuf + (size_t)bb * Sc * Dc;
  const bf16_t* vbase = vt   + (size_t)bb * Dc * Sc;

  auto stage = [&](int buf, int kb) {
    int key = tid >> 3, seg = tid & 7;                 // 32 keys x 128 d
    const bf16_t* gk = kbase + (size_t)(kb + key) * Dc + seg * 16;
    bf16_t* lk = &kt[buf][key * 128 + seg * 16];
    int dim = tid >> 1, vseg = tid & 1;                // 128 d x 32 keys
    const bf16_t* gv = vbase + (size_t)dim * Sc + kb + vseg * 16;
    bf16_t* lv = &vtile[buf][dim * 32 + vseg * 16];
#if USE_ASYNC_LDS
    async_cp16(gk, lk); async_cp16(gk + 8, lk + 8);
    async_cp16(gv, lv); async_cp16(gv + 8, lv + 8);
#else
    copy32B(lk, gk);
    copy32B(lv, gv);
#endif
  };
  auto stage_fence = []() {
#if USE_ASYNC_LDS
    __builtin_amdgcn_s_wait_asynccnt(0);   // this wave's async copies landed
#endif
  };

  // Q fragments for all 4 K-chunks of D=128, resident in registers
  const bf16_t* qrow = qb + ((size_t)((bb * Hc + h) * Sc + q0 + l16)) * Dc;
  v16bf qf[4];
  #pragma unroll
  for (int dc = 0; dc < 4; ++dc)
    qf[dc] = load8x2(qrow + dc * 32 + 16 * half, qrow + dc * 32 + 16 * half + 8);

  v8f z = {};
  v8f oacc[8];
  #pragma unroll
  for (int n = 0; n < 8; ++n) oacc[n] = z;
  float runm[8], runl[8];
  #pragma unroll
  for (int r = 0; r < 8; ++r) { runm[r] = -1e30f; runl[r] = 0.f; }

  stage(0, 0);
  stage_fence();
  __syncthreads();

  for (int kb = 0; kb < Sc; kb += 32) {
    const int cur = (kb >> 5) & 1;
    if (kb + 32 < Sc) stage(cur ^ 1, kb + 32);         // overlap next-block staging

    v8f s0 = z, s1 = z;
    #pragma unroll
    for (int dc = 0; dc < 4; ++dc) {  // B-fragment: lane = key column, elems = head dim
      v16bf k0 = load16(&kt[cur][l16 * 128 + dc * 32 + 16 * half]);
      v16bf k1 = load16(&kt[cur][(16 + l16) * 128 + dc * 32 + 16 * half]);
      s0 = wmma_bf16(qf[dc], k0, s0);
      s1 = wmma_bf16(qf[dc], k1, s1);
    }
    #pragma unroll
    for (int r = 0; r < 8; ++r) { s0[r] *= scale; s1[r] *= scale; }

    float p0[8], p1[8], alpha[8];
    #pragma unroll
    for (int r = 0; r < 8; ++r) {      // row r+8*half lives in this half-wave
      float m = fmaxf(s0[r], s1[r]);
      #pragma unroll
      for (int d = 8; d >= 1; d >>= 1) m = fmaxf(m, __shfl_xor(m, d, 16));
      float nm = fmaxf(runm[r], m);
      alpha[r] = __expf(runm[r] - nm);
      runm[r] = nm;
      p0[r] = __expf(s0[r] - nm);
      p1[r] = __expf(s1[r] - nm);
      float rs = p0[r] + p1[r];
      #pragma unroll
      for (int d = 8; d >= 1; d >>= 1) rs += __shfl_xor(rs, d, 16);
      runl[r] = runl[r] * alpha[r] + rs;
    }
    #pragma unroll
    for (int n = 0; n < 8; ++n)
      #pragma unroll
      for (int r = 0; r < 8; ++r) oacc[n][r] *= alpha[r];

    // Transpose P from C-layout to A-layout through per-wave LDS (in-order DS)
    #pragma unroll
    for (int r = 0; r < 8; ++r) {
      int row = r + 8 * half;
      pw[row * 32 + l16]      = f2bf(p0[r]);
      pw[row * 32 + 16 + l16] = f2bf(p1[r]);
    }
    v16bf pf = load8x2(&pw[l16 * 32 + 16 * half], &pw[l16 * 32 + 16 * half + 8]);
    #pragma unroll
    for (int n = 0; n < 8; ++n) {      // B-fragment from V^T: lane = out dim, elems = keys
      v16bf vf = load16(&vtile[cur][(n * 16 + l16) * 32 + 16 * half]);
      oacc[n] = wmma_bf16(pf, vf, oacc[n]);
    }
    stage_fence();                     // staged next buffer landed (this wave)
    __syncthreads();                   // publish to all waves
  }

  float inv[8];
  #pragma unroll
  for (int r = 0; r < 8; ++r) inv[r] = 1.f / runl[r];
  #pragma unroll
  for (int n = 0; n < 8; ++n)
    #pragma unroll
    for (int r = 0; r < 8; ++r) {
      int q = q0 + r + 8 * half;
      ao[((size_t)(bb * Sc + q)) * (Hc * Dc) + h * Dc + n * 16 + l16] =
          f2bf(oacc[n][r] * inv[r]);
    }
}

// ---------------------------------------------------------------------------
// Kernel 3: output projection  ao(bf16) @ woT(bf16, n-major) -> out(f32).
// Wave tile 32x64, register double-buffered; no LDS, no barriers.
// ---------------------------------------------------------------------------
__global__ __launch_bounds__(128)
void oproj_kernel(const bf16_t* __restrict__ ao, const bf16_t* __restrict__ woT,
                  float* __restrict__ out)
{
  const int tid = threadIdx.x, lane = tid & 31, wave = tid >> 5;
  const int half = lane >> 4, l16 = lane & 15;
  const int mi = wave >> 1, ni = wave & 1;
  const int m0 = blockIdx.x * 64, n0 = blockIdx.y * 128;

  v8f z = {};
  v8f acc[2][4];
  #pragma unroll
  for (int m = 0; m < 2; ++m)
    #pragma unroll
    for (int j = 0; j < 4; ++j) acc[m][j] = z;

  const bf16_t* arow[2];
  #pragma unroll
  for (int m = 0; m < 2; ++m)
    arow[m] = ao + (size_t)(m0 + mi * 32 + m * 16 + l16) * DIMc;
  const bf16_t* bcol[4];
  #pragma unroll
  for (int j = 0; j < 4; ++j)
    bcol[j] = woT + (size_t)(n0 + ni * 64 + j * 16 + l16) * DIMc + 16 * half;

  auto loadA = [&](v16bf af[2], int kb) {
    #pragma unroll
    for (int m = 0; m < 2; ++m)
      af[m] = load8x2(arow[m] + kb + 16 * half, arow[m] + kb + 16 * half + 8);
  };
  auto loadB = [&](v16bf bfv[4], int kb) {
    #pragma unroll
    for (int j = 0; j < 4; ++j) bfv[j] = load16(bcol[j] + kb);
  };
  auto mmas = [&](v16bf af[2], v16bf bfv[4]) {
    #pragma unroll
    for (int m = 0; m < 2; ++m)
      #pragma unroll
      for (int j = 0; j < 4; ++j)
        acc[m][j] = wmma_bf16(af[m], bfv[j], acc[m][j]);
  };

  v16bf a0[2], b0[4], a1[2], b1[4];
  loadA(a0, 0); loadB(b0, 0);
  for (int kb = 0; kb < DIMc; kb += 64) {
    loadA(a1, kb + 32); loadB(b1, kb + 32);
    mmas(a0, b0);
    if (kb + 64 < DIMc) { loadA(a0, kb + 64); loadB(b0, kb + 64); }
    mmas(a1, b1);
  }

  #pragma unroll
  for (int m = 0; m < 2; ++m)
    #pragma unroll
    for (int j = 0; j < 4; ++j)
      #pragma unroll
      for (int r = 0; r < 8; ++r)
        out[(size_t)(m0 + mi * 32 + m * 16 + r + 8 * half) * DIMc +
            n0 + ni * 64 + j * 16 + l16] = acc[m][j][r];
}

// ---------------------------------------------------------------------------
extern "C" void kernel_launch(void* const* d_in, const int* in_sizes, int n_in,
                              void* d_out, int out_size, void* d_ws, size_t ws_size,
                              hipStream_t stream) {
  const float* hidden = (const float*)d_in[0];
  const float* cosv   = (const float*)d_in[1];
  const float* sinv   = (const float*)d_in[2];
  const float* w_q    = (const float*)d_in[3];
  const float* w_k    = (const float*)d_in[4];
  const float* w_v    = (const float*)d_in[5];
  const float* w_o    = (const float*)d_in[6];
  const float* ln_q_w = (const float*)d_in[7];
  const float* ln_q_b = (const float*)d_in[8];
  const float* ln_k_w = (const float*)d_in[9];
  const float* ln_k_b = (const float*)d_in[10];
  float* out = (float*)d_out;

  // workspace (all bf16): ~117 MB total
  bf16_t* qb   = (bf16_t*)d_ws;                       // [2][24][2048][128]
  bf16_t* kbuf = qb   + (size_t)Bc * Hc * Sc * Dc;    // [2][2048][128]
  bf16_t* vt   = kbuf + (size_t)Bc * Sc * Dc;         // [2][128][2048]
  bf16_t* ao   = vt   + (size_t)Bc * Sc * Dc;         // [2][2048][3072]
  bf16_t* hb   = ao   + (size_t)Bc * Sc * Hc * Dc;    // [2*2048][3072]
  bf16_t* wqT  = hb   + (size_t)Bc * Sc * DIMc;       // [3072][3072]
  bf16_t* wkT  = wqT  + (size_t)DIMc * DIMc;          // [128][3072]
  bf16_t* wvT  = wkT  + (size_t)Dc * DIMc;            // [128][3072]
  bf16_t* woT  = wvT  + (size_t)Dc * DIMc;            // [3072][3072]

  // one-time convert / transpose pre-passes
  int n4 = (Bc * Sc * DIMc) / 4;
  convert_bf16_kernel<<<dim3((n4 + 255) / 256), 256, 0, stream>>>(hidden, hb, n4);
  transpose_convert_kernel<<<dim3(DIMc / 32, DIMc / 32), 256, 0, stream>>>(w_q, wqT, DIMc, DIMc);
  transpose_convert_kernel<<<dim3(Dc / 32,   DIMc / 32), 256, 0, stream>>>(w_k, wkT, DIMc, Dc);
  transpose_convert_kernel<<<dim3(Dc / 32,   DIMc / 32), 256, 0, stream>>>(w_v, wvT, DIMc, Dc);
  transpose_convert_kernel<<<dim3(DIMc / 32, DIMc / 32), 256, 0, stream>>>(w_o, woT, DIMc, DIMc);

  qkv_ln_rope_kernel<<<dim3(64, 26), 128, 0, stream>>>(
      hb, wqT, wkT, wvT, cosv, sinv, ln_q_w, ln_q_b, ln_k_w, ln_k_b,
      qb, kbuf, vt);
  attention_kernel<<<dim3(16, 24, 2), 256, 0, stream>>>(qb, kbuf, vt, ao);
  oproj_kernel<<<dim3(64, 24), 128, 0, stream>>>(ao, woT, out);

  (void)in_sizes; (void)n_in; (void)out_size; (void)ws_size;
}
